// MeanPooledSpanEmbeddingLayer_87316685128607
// MI455X (gfx1250) — matrix operand order repaired and checked by reading, not compiled, not run on hardware
//
#include <hip/hip_runtime.h>
#include <hip/hip_bf16.h>
#include <math.h>

// ---------------------------------------------------------------------------
// MI455X (gfx1250): all GEMMs on v_wmma_f32_16x16x32_bf16 (wave32). Weights are
// pre-converted to bf16 once per launch (they live in the 192MB L2). The two
// biggest per-token GEMMs stage their A-tiles into LDS with the Tensor Data
// Mover (tensor_load_to_lds, double-buffered on TENSORcnt).
// ---------------------------------------------------------------------------

typedef __bf16 bf16;
typedef __attribute__((ext_vector_type(16))) __bf16 bf16x16;
typedef __attribute__((ext_vector_type(8)))  __bf16 bf16x8;
typedef __attribute__((ext_vector_type(8)))  float  f32x8;
typedef __attribute__((ext_vector_type(4)))  unsigned int u32x4;
typedef __attribute__((ext_vector_type(8)))  int i32x8;
typedef __attribute__((ext_vector_type(4)))  int i32x4;

constexpr int kB  = 4;
constexpr int kS  = 4096;
constexpr int kH  = 1536;
constexpr int kH2 = 768;
constexpr int kNS = 512;
constexpr int kNT = 55;
constexpr int kM  = kB * kS;    // 16384 token rows
constexpr int kMS = kB * kNS;   // 2048 segments
constexpr int k2H = 2 * kH;     // 3072
constexpr int kSlab = 512;      // K-slab staged via TDM

#if defined(__has_builtin)
#if __has_builtin(__builtin_amdgcn_tensor_load_to_lds) && \
    __has_builtin(__builtin_amdgcn_s_wait_tensorcnt)
#define USE_TDM 1
#endif
#endif
#ifndef USE_TDM
#define USE_TDM 0
#endif

// ---------------- TDM 2D tile load: global bf16 tile -> LDS ----------------
// D# group0: count=1, lds_addr, 57-bit global addr, type=2. group1: data_size=2B,
// tensor_dim0/1, tile_dim0/1, dim0 stride. Groups 2/3/(pad) zero (<=2D tensor).
__device__ __forceinline__ void tdm_load_2d_bf16(const bf16* gptr, unsigned lds_off,
                                                 unsigned tensor_d0, unsigned tensor_d1,
                                                 unsigned tile_d0, unsigned tile_d1,
                                                 unsigned stride0_elems) {
#if USE_TDM
  unsigned long long ga = (unsigned long long)gptr;
  u32x4 g0;
  g0[0] = 1u;                                                  // count=1, no gather
  g0[1] = lds_off;                                             // LDS byte address
  g0[2] = (unsigned)ga;                                        // global addr [31:0]
  g0[3] = (unsigned)((ga >> 32) & 0x01FFFFFFu) | 0x80000000u;  // addr[56:32] | type=2
  i32x8 g1;
  g1[0] = (int)(1u << 16);                                     // data_size = 2 bytes
  g1[1] = (int)((tensor_d0 & 0xFFFFu) << 16);                  // tensor_dim0[15:0]
  g1[2] = (int)((tensor_d0 >> 16) | ((tensor_d1 & 0xFFFFu) << 16));
  g1[3] = (int)((tensor_d1 >> 16) | (tile_d0 << 16));          // tile_dim0
  g1[4] = (int)(tile_d1 & 0xFFFFu);                            // tile_dim1 (tile_dim2=0)
  g1[5] = (int)stride0_elems;                                  // tensor_dim0_stride[31:0]
  g1[6] = 0;
  g1[7] = 0;
  i32x4 z4 = {0, 0, 0, 0};
  i32x8 z8 = {0, 0, 0, 0, 0, 0, 0, 0};
  __builtin_amdgcn_tensor_load_to_lds(g0, g1, z4, z4, z8, 0);
#else
  (void)gptr; (void)lds_off; (void)tensor_d0; (void)tensor_d1;
  (void)tile_d0; (void)tile_d1; (void)stride0_elems;
#endif
}

// ---------------- WMMA fragment helpers (wave32) ----------------
// A 16x32 bf16: lane<16 -> M=lane, h=lane>>4; elems 0..7 = K 8h+0..7, 8..15 = 16+8h+0..7.
__device__ __forceinline__ bf16x16 frag_a_bf16(const bf16* A, int lda, int m0, int k0) {
  int l = threadIdx.x & 31;
  int m = m0 + (l & 15);
  int h = l >> 4;
  const bf16* p = A + (size_t)m * lda + k0 + 8 * h;
  bf16x8 lo = *(const bf16x8*)p;
  bf16x8 hi = *(const bf16x8*)(p + 16);
  bf16x16 f;
#pragma unroll
  for (int e = 0; e < 8; ++e) { f[e] = lo[e]; f[8 + e] = hi[e]; }
  return f;
}

// B 32x16 bf16 from bf16 W [N,K] row-major: lane -> N=l&15, elems 0..7 -> K=8h+e,
// 8..15 -> K=16+8h+(e-8) (two b128 loads).
__device__ __forceinline__ bf16x16 frag_b_bf16(const bf16* W, int ldw, int n0, int k0) {
  int l = threadIdx.x & 31;
  int n = n0 + (l & 15);
  int h = l >> 4;
  const bf16* p = W + (size_t)n * ldw + k0 + 8 * h;
  bf16x8 lo = *(const bf16x8*)p;
  bf16x8 hi = *(const bf16x8*)(p + 16);
  bf16x16 f;
#pragma unroll
  for (int e = 0; e < 8; ++e) { f[e] = lo[e]; f[8 + e] = hi[e]; }
  return f;
}

__device__ __forceinline__ f32x8 wmma_bf16(bf16x16 a, bf16x16 b, f32x8 c) {
  return __builtin_amdgcn_wmma_f32_16x16x32_bf16(false, a, false, b,
                                                 (short)0, c, false, false);
}

__device__ __forceinline__ float block_sum(float v, float* red) {
  int tid = threadIdx.x;
  red[tid] = v;
  __syncthreads();
#pragma unroll
  for (int s = 128; s > 0; s >>= 1) {
    if (tid < s) red[tid] += red[tid + s];
    __syncthreads();
  }
  float r = red[0];
  __syncthreads();
  return r;
}

// ---------------- weight pre-conversion ----------------
__global__ void convert_bf16_kernel(const float* __restrict__ src,
                                    bf16* __restrict__ dst, int n) {
  int i = blockIdx.x * 256 + threadIdx.x;
  if (i < n) dst[i] = (bf16)src[i];
}

__global__ void convert_hw2_kernel(const float* __restrict__ hw2,
                                   bf16* __restrict__ hw2b) {
  int i = blockIdx.x * 256 + threadIdx.x;  // over 64*768 (zero-padded rows 55..63)
  if (i < 64 * kH2) {
    int n = i / kH2, k = i - n * kH2;
    hw2b[i] = (bf16)(n < kNT ? hw2[n * kH2 + k] : 0.0f);
  }
}

// ---------------- Kernel A: embedding gather + residual adapter ----------------
__global__ __launch_bounds__(256) void adapter_kernel(
    const int* __restrict__ ids, const float* __restrict__ emb,
    const bf16* __restrict__ aw1b, const float* __restrict__ ab1,
    const bf16* __restrict__ aw2b, const float* __restrict__ ab2,
    float* __restrict__ tok) {
  __shared__ __align__(16) bf16 sA[16 * kH];
  __shared__ __align__(16) bf16 sH1[16 * kH];
  int m0 = blockIdx.x * 16;

  for (int idx = threadIdx.x; idx < 16 * kH; idx += 256) {
    int r = idx / kH, c = idx - r * kH;
    sA[idx] = (bf16)emb[(size_t)ids[m0 + r] * kH + c];
  }
  __syncthreads();

  int wave = threadIdx.x >> 5;
  int l = threadIdx.x & 31, nl = l & 15, h = l >> 4;

  for (int nt = wave; nt < kH / 16; nt += 8) {
    int n0 = nt * 16;
    f32x8 acc = {};
    for (int k0 = 0; k0 < kH; k0 += 32)
      acc = wmma_bf16(frag_a_bf16(sA, kH, 0, k0), frag_b_bf16(aw1b, kH, n0, k0), acc);
    float bias = ab1[n0 + nl];
#pragma unroll
    for (int r = 0; r < 8; ++r) {
      float x = acc[r] + bias;
      x = 0.5f * x * (1.0f + erff(x * 0.70710678118f));   // exact GELU
      sH1[(r + 8 * h) * kH + (n0 + nl)] = (bf16)x;
    }
  }
  __syncthreads();

  for (int nt = wave; nt < kH / 16; nt += 8) {
    int n0 = nt * 16;
    f32x8 acc = {};
    for (int k0 = 0; k0 < kH; k0 += 32)
      acc = wmma_bf16(frag_a_bf16(sH1, kH, 0, k0), frag_b_bf16(aw2b, kH, n0, k0), acc);
    float bias = ab2[n0 + nl];
#pragma unroll
    for (int r = 0; r < 8; ++r) {
      int row = m0 + r + 8 * h;
      int col = n0 + nl;
      float res = emb[(size_t)ids[row] * kH + col];
      tok[(size_t)row * kH + col] = acc[r] + bias + res;
    }
  }
}

// ---------------- Pooling ----------------
__global__ void zero_kernel(float* __restrict__ p, int n) {
  int i = blockIdx.x * 256 + threadIdx.x;
  if (i < n) p[i] = 0.0f;
}

__global__ void pool_count_kernel(const int* __restrict__ span_ids,
                                  float* __restrict__ counts) {
  int t = blockIdx.x * 256 + threadIdx.x;
  if (t < kM) {
    int seg = (t / kS) * kNS + span_ids[t];
    atomicAdd(&counts[seg], 1.0f);
  }
}

__global__ __launch_bounds__(384) void pool_sum_kernel(
    const int* __restrict__ span_ids, const float* __restrict__ tok,
    float* __restrict__ sums) {
  int t = blockIdx.x;
  int seg = (t / kS) * kNS + span_ids[t];
  const float4* src = (const float4*)(tok + (size_t)t * kH);
  float* dst = sums + (size_t)seg * kH;
  int c = threadIdx.x;
  float4 v = src[c];
  atomicAdd(&dst[4 * c + 0], v.x);
  atomicAdd(&dst[4 * c + 1], v.y);
  atomicAdd(&dst[4 * c + 2], v.z);
  atomicAdd(&dst[4 * c + 3], v.w);
}

// ---------------- Kernel C: enhanced = (pooled + type_emb) @ pw.T + pb ----------------
__global__ __launch_bounds__(256) void enhanced_kernel(
    const int* __restrict__ span_types, const float* __restrict__ sums,
    const float* __restrict__ counts, const float* __restrict__ type_table,
    const bf16* __restrict__ pwb, const float* __restrict__ pb,
    float* __restrict__ enhanced) {
  __shared__ __align__(16) bf16 sA[16 * kH];
  int m0 = blockIdx.x * 16;
  for (int idx = threadIdx.x; idx < 16 * kH; idx += 256) {
    int r = idx / kH, c = idx - r * kH;
    int m = m0 + r;
    float p = sums[(size_t)m * kH + c] / fmaxf(counts[m], 1.0f);
    sA[idx] = (bf16)(p + type_table[(size_t)span_types[m] * kH + c]);
  }
  __syncthreads();

  int wave = threadIdx.x >> 5;
  int l = threadIdx.x & 31, nl = l & 15, h = l >> 4;
  for (int nt = wave; nt < kH / 16; nt += 8) {
    int n0 = nt * 16;
    f32x8 acc = {};
    for (int k0 = 0; k0 < kH; k0 += 32)
      acc = wmma_bf16(frag_a_bf16(sA, kH, 0, k0), frag_b_bf16(pwb, kH, n0, k0), acc);
    float bias = pb[n0 + nl];
#pragma unroll
    for (int r = 0; r < 8; ++r)
      enhanced[(size_t)(m0 + r + 8 * h) * kH + (n0 + nl)] = acc[r] + bias;
  }
}

// ---------------- Kernel D: span_emb gather + LN -> gate_in (bf16 [M,2H]) ----------------
__global__ __launch_bounds__(256) void gate_prep_kernel(
    const int* __restrict__ span_ids, const float* __restrict__ tok,
    const float* __restrict__ enhanced, const float* __restrict__ counts,
    const float* __restrict__ snw, const float* __restrict__ snb,
    bf16* __restrict__ gate_in, float* __restrict__ span_mask) {
  __shared__ float red[256];
  int t = blockIdx.x;
  int seg = (t / kS) * kNS + span_ids[t];
  bool valid = counts[seg] >= 2.0f;
  const float* src = valid ? (enhanced + (size_t)seg * kH) : (tok + (size_t)t * kH);
  int tid = threadIdx.x;

  float s = 0.0f;
  for (int c = tid; c < kH; c += 256) s += src[c];
  float mean = block_sum(s, red) * (1.0f / kH);
  float v = 0.0f;
  for (int c = tid; c < kH; c += 256) { float d = src[c] - mean; v += d * d; }
  float rstd = rsqrtf(block_sum(v, red) * (1.0f / kH) + 1e-5f);

  bf16* dst = gate_in + (size_t)t * k2H;
  for (int c = tid; c < kH; c += 256) {
    dst[c]      = (bf16)tok[(size_t)t * kH + c];
    dst[kH + c] = (bf16)((src[c] - mean) * rstd * snw[c] + snb[c]);
  }
  if (tid == 0) span_mask[t] = valid ? 1.0f : 0.0f;
}

// ---------------- Kernel E: gate GEMM (A staged via TDM, double-buffered) ----------------
// grid = (kM/16, 4): block does 16 rows x 24 N-tiles; wave owns 3 N-tiles.
__global__ __launch_bounds__(256) void gate_gemm_kernel(
    const bf16* __restrict__ gate_in, const bf16* __restrict__ gwb,
    const float* __restrict__ gb, const float* __restrict__ span_mask,
    float* __restrict__ gate) {
  __shared__ __align__(16) bf16 sA[2][16 * kSlab];
  int m0 = blockIdx.x * 16;
  int nt0 = blockIdx.y * 24;
  int wave = threadIdx.x >> 5;
  int l = threadIdx.x & 31, nl = l & 15, h = l >> 4;
  constexpr int kSlabs = k2H / kSlab;  // 6
  const bf16* abase = gate_in + (size_t)m0 * k2H;

  f32x8 acc[3] = {};

#if USE_TDM
  if (wave == 0)
    tdm_load_2d_bf16(abase, (unsigned)(size_t)&sA[0][0], k2H, 16, kSlab, 16, k2H);
#endif
  for (int s = 0; s < kSlabs; ++s) {
    int buf = s & 1;
#if USE_TDM
    if (wave == 0) {
      if (s + 1 < kSlabs) {
        tdm_load_2d_bf16(abase + (s + 1) * kSlab, (unsigned)(size_t)&sA[buf ^ 1][0],
                         k2H, 16, kSlab, 16, k2H);
        __builtin_amdgcn_s_wait_tensorcnt(1);  // slab s complete, s+1 in flight
      } else {
        __builtin_amdgcn_s_wait_tensorcnt(0);
      }
    }
#else
    for (int idx = threadIdx.x; idx < 16 * kSlab / 8; idx += 256) {
      int r = idx / (kSlab / 8), c8 = idx - r * (kSlab / 8);
      ((bf16x8*)sA[buf])[idx] =
          *(const bf16x8*)(abase + (size_t)r * k2H + s * kSlab + c8 * 8);
    }
#endif
    if (l == 0 && s + 1 < kSlabs)
      __builtin_prefetch(gwb + (size_t)(nt0 + wave * 3) * 16 * k2H + (s + 1) * kSlab, 0, 1);
    __syncthreads();
    for (int k0 = 0; k0 < kSlab; k0 += 32) {
      bf16x16 a = frag_a_bf16(sA[buf], kSlab, 0, k0);
#pragma unroll
      for (int j = 0; j < 3; ++j)
        acc[j] = wmma_bf16(a, frag_b_bf16(gwb, k2H, (nt0 + wave * 3 + j) * 16,
                                          s * kSlab + k0), acc[j]);
    }
    __syncthreads();
  }

#pragma unroll
  for (int j = 0; j < 3; ++j) {
    int n0 = (nt0 + wave * 3 + j) * 16;
    float bias = gb[n0 + nl];
#pragma unroll
    for (int r = 0; r < 8; ++r) {
      int row = m0 + r + 8 * h;
      float g = 1.0f / (1.0f + __expf(-(acc[j][r] + bias)));
      gate[(size_t)row * kH + (n0 + nl)] = g * span_mask[row];
    }
  }
}

// ---------------- Kernel F: combined + LN -> logits_in (bf16) ----------------
__global__ __launch_bounds__(256) void combine_kernel(
    const int* __restrict__ span_ids, const float* __restrict__ tok,
    const float* __restrict__ enhanced, const float* __restrict__ counts,
    const float* __restrict__ gate, const float* __restrict__ lnw,
    const float* __restrict__ lnb, float* __restrict__ combined,
    bf16* __restrict__ logits_in) {
  __shared__ float red[256];
  __shared__ float sRow[kH];
  int t = blockIdx.x;
  int seg = (t / kS) * kNS + span_ids[t];
  bool valid = counts[seg] >= 2.0f;
  const float* sp = valid ? (enhanced + (size_t)seg * kH) : (tok + (size_t)t * kH);
  int tid = threadIdx.x;

  float s = 0.0f;
  for (int c = tid; c < kH; c += 256) {
    float g = gate[(size_t)t * kH + c];
    float x = (1.0f - g) * tok[(size_t)t * kH + c] + g * sp[c];
    combined[(size_t)t * kH + c] = x;
    sRow[c] = x;
    s += x;
  }
  float mean = block_sum(s, red) * (1.0f / kH);
  float v = 0.0f;
  for (int c = tid; c < kH; c += 256) { float d = sRow[c] - mean; v += d * d; }
  float rstd = rsqrtf(block_sum(v, red) * (1.0f / kH) + 1e-5f);
  for (int c = tid; c < kH; c += 256)
    logits_in[(size_t)t * kH + c] = (bf16)((sRow[c] - mean) * rstd * lnw[c] + lnb[c]);
}

// ---------------- Kernel G: head (relu 1536->768 staged via TDM, then 768->55) ----------------
__global__ __launch_bounds__(256) void head_kernel(
    const bf16* __restrict__ logits_in, const bf16* __restrict__ hw1b,
    const float* __restrict__ hb1, const bf16* __restrict__ hw2b,
    const float* __restrict__ hb2, float* __restrict__ logits) {
  __shared__ __align__(16) bf16 sA[2][16 * kSlab];
  __shared__ __align__(16) bf16 sH2[16 * kH2];
  int m0 = blockIdx.x * 16;
  int wave = threadIdx.x >> 5;
  int l = threadIdx.x & 31, nl = l & 15, h = l >> 4;
  constexpr int kSlabs = kH / kSlab;  // 3
  const bf16* abase = logits_in + (size_t)m0 * kH;

  f32x8 acc[6] = {};

#if USE_TDM
  if (wave == 0)
    tdm_load_2d_bf16(abase, (unsigned)(size_t)&sA[0][0], kH, 16, kSlab, 16, kH);
#endif
  for (int s = 0; s < kSlabs; ++s) {
    int buf = s & 1;
#if USE_TDM
    if (wave == 0) {
      if (s + 1 < kSlabs) {
        tdm_load_2d_bf16(abase + (s + 1) * kSlab, (unsigned)(size_t)&sA[buf ^ 1][0],
                         kH, 16, kSlab, 16, kH);
        __builtin_amdgcn_s_wait_tensorcnt(1);
      } else {
        __builtin_amdgcn_s_wait_tensorcnt(0);
      }
    }
#else
    for (int idx = threadIdx.x; idx < 16 * kSlab / 8; idx += 256) {
      int r = idx / (kSlab / 8), c8 = idx - r * (kSlab / 8);
      ((bf16x8*)sA[buf])[idx] =
          *(const bf16x8*)(abase + (size_t)r * kH + s * kSlab + c8 * 8);
    }
#endif
    if (l == 0 && s + 1 < kSlabs)
      __builtin_prefetch(hw1b + (size_t)(wave * 6) * 16 * kH + (s + 1) * kSlab, 0, 1);
    __syncthreads();
    for (int k0 = 0; k0 < kSlab; k0 += 32) {
      bf16x16 a = frag_a_bf16(sA[buf], kSlab, 0, k0);
#pragma unroll
      for (int j = 0; j < 6; ++j)
        acc[j] = wmma_bf16(a, frag_b_bf16(hw1b, kH, (wave * 6 + j) * 16,
                                          s * kSlab + k0), acc[j]);
    }
    __syncthreads();
  }

#pragma unroll
  for (int j = 0; j < 6; ++j) {
    int n0 = (wave * 6 + j) * 16;
    float bias = hb1[n0 + nl];
#pragma unroll
    for (int r = 0; r < 8; ++r)
      sH2[(r + 8 * h) * kH2 + (n0 + nl)] = (bf16)fmaxf(acc[j][r] + bias, 0.0f);
  }
  __syncthreads();

  if (wave < 4) {  // 4 N-tiles cover NT=55 (bf16 weights zero-padded to 64 rows)
    int n0 = wave * 16;
    f32x8 a2 = {};
    for (int k0 = 0; k0 < kH2; k0 += 32)
      a2 = wmma_bf16(frag_a_bf16(sH2, kH2, 0, k0),
                     frag_b_bf16(hw2b, kH2, n0, k0), a2);
#pragma unroll
    for (int r = 0; r < 8; ++r) {
      int n = n0 + nl;
      if (n < kNT)
        logits[(size_t)(m0 + r + 8 * h) * kNT + n] = a2[r] + hb2[n];
    }
  }
}

// ---------------- Host launch ----------------
extern "C" void kernel_launch(void* const* d_in, const int* in_sizes, int n_in,
                              void* d_out, int out_size, void* d_ws, size_t ws_size,
                              hipStream_t stream) {
  (void)in_sizes; (void)n_in; (void)out_size; (void)ws_size;

  const int*   input_ids  = (const int*)  d_in[0];
  const int*   span_ids   = (const int*)  d_in[1];
  const int*   span_types = (const int*)  d_in[2];
  const float* emb_table  = (const float*)d_in[3];
  const float* aw1 = (const float*)d_in[4];
  const float* ab1 = (const float*)d_in[5];
  const float* aw2 = (const float*)d_in[6];
  const float* ab2 = (const float*)d_in[7];
  const float* type_table = (const float*)d_in[8];
  const float* pw  = (const float*)d_in[9];
  const float* pb  = (const float*)d_in[10];
  const float* snw = (const float*)d_in[11];
  const float* snb = (const float*)d_in[12];
  const float* gw  = (const float*)d_in[13];
  const float* gb  = (const float*)d_in[14];
  const float* lnw = (const float*)d_in[15];
  const float* lnb = (const float*)d_in[16];
  const float* hw1 = (const float*)d_in[17];
  const float* hb1 = (const float*)d_in[18];
  const float* hw2 = (const float*)d_in[19];
  const float* hb2 = (const float*)d_in[20];

  // workspace layout
  float* tok       = (float*)d_ws;                          // [M,H] fp32
  float* sums      = tok + (size_t)kM * kH;                 // [MS,H] fp32
  float* counts    = sums + (size_t)kMS * kH;               // [MS] fp32
  float* enhanced  = counts + kMS;                          // [MS,H] fp32
  bf16*  gate_in   = (bf16*)(enhanced + (size_t)kMS * kH);  // [M,2H] bf16
  bf16*  logits_in = gate_in + (size_t)kM * k2H;            // [M,H] bf16
  bf16*  aw1b      = logits_in + (size_t)kM * kH;           // bf16 weights
  bf16*  aw2b      = aw1b + (size_t)kH * kH;
  bf16*  pwb       = aw2b + (size_t)kH * kH;
  bf16*  gwb       = pwb  + (size_t)kH * kH;
  bf16*  hw1b      = gwb  + (size_t)kH * k2H;
  bf16*  hw2b      = hw1b + (size_t)kH2 * kH;               // [64,768] padded

  // output layout: combined, span_type_logits, gate, span_mask
  float* out      = (float*)d_out;
  float* combined = out;
  float* logits   = combined + (size_t)kM * kH;
  float* gate     = logits + (size_t)kM * kNT;
  float* mask     = gate + (size_t)kM * kH;

  // weight pre-conversion (weights stay hot in 192MB L2 afterwards)
  int nw = kH * kH;
  convert_bf16_kernel<<<(nw + 255) / 256, 256, 0, stream>>>(aw1, aw1b, nw);
  convert_bf16_kernel<<<(nw + 255) / 256, 256, 0, stream>>>(aw2, aw2b, nw);
  convert_bf16_kernel<<<(nw + 255) / 256, 256, 0, stream>>>(pw, pwb, nw);
  int ngw = kH * k2H;
  convert_bf16_kernel<<<(ngw + 255) / 256, 256, 0, stream>>>(gw, gwb, ngw);
  int nh1 = kH2 * kH;
  convert_bf16_kernel<<<(nh1 + 255) / 256, 256, 0, stream>>>(hw1, hw1b, nh1);
  int nh2 = 64 * kH2;
  convert_hw2_kernel<<<(nh2 + 255) / 256, 256, 0, stream>>>(hw2, hw2b);

  adapter_kernel<<<kM / 16, 256, 0, stream>>>(input_ids, emb_table, aw1b, ab1,
                                              aw2b, ab2, tok);
  int nzero = kMS * kH + kMS;
  zero_kernel<<<(nzero + 255) / 256, 256, 0, stream>>>(sums, nzero);
  pool_count_kernel<<<(kM + 255) / 256, 256, 0, stream>>>(span_ids, counts);
  pool_sum_kernel<<<kM, 384, 0, stream>>>(span_ids, tok, sums);
  enhanced_kernel<<<kMS / 16, 256, 0, stream>>>(span_types, sums, counts,
                                                type_table, pwb, pb, enhanced);
  gate_prep_kernel<<<kM, 256, 0, stream>>>(span_ids, tok, enhanced, counts,
                                           snw, snb, gate_in, mask);
  gate_gemm_kernel<<<dim3(kM / 16, 4), 256, 0, stream>>>(gate_in, gwb, gb, mask, gate);
  combine_kernel<<<kM, 256, 0, stream>>>(span_ids, tok, enhanced, counts, gate,
                                         lnw, lnb, combined, logits_in);
  head_kernel<<<kM / 16, 256, 0, stream>>>(logits_in, hw1b, hb1, hw2b, hb2, logits);
}
